// ChameleonVQVAEVectorQuantizer_47562467836327
// MI455X (gfx1250) — matrix operand order repaired and checked by reading, not compiled, not run on hardware
//
#include <hip/hip_runtime.h>

// ---------------------------------------------------------------------------
// Problem constants (from reference): hidden [16,256,32,32] f32, emb [8192,256]
// ---------------------------------------------------------------------------
#define NROWS 16384   // 16 * 32 * 32
#define CDIM  256
#define KCODES 8192
#define HWSZ  1024    // 32*32

typedef __attribute__((ext_vector_type(8)))  float  v8f;
typedef __attribute__((ext_vector_type(16))) __bf16 v16bf;

union BfVec { v16bf v; uint4 q[2]; };

// ---------------------------------------------------------------------------
// bf16 helpers (round-to-nearest-even) for hi/lo split of fp32
// ---------------------------------------------------------------------------
__device__ __forceinline__ unsigned short f32_to_bf16_rne(float f) {
    unsigned int u = __float_as_uint(f);
    unsigned int r = 0x7FFFu + ((u >> 16) & 1u);
    u += r;
    return (unsigned short)(u >> 16);
}
__device__ __forceinline__ float bf16_to_f32(unsigned short h) {
    return __uint_as_float(((unsigned int)h) << 16);
}

// ---------------------------------------------------------------------------
// Prep: codebook -> bf16 hi/lo planes + exact ||e||^2
// grid = KCODES blocks, 256 threads (one element per thread)
// ---------------------------------------------------------------------------
__global__ void __launch_bounds__(256)
vq_prep_codes(const float* __restrict__ emb,
              unsigned short* __restrict__ Ehi,
              unsigned short* __restrict__ Elo,
              float* __restrict__ esq) {
    __shared__ float s[8];
    const int k = blockIdx.x;
    const int t = threadIdx.x;
    const float v = emb[k * CDIM + t];
    const unsigned short hi = f32_to_bf16_rne(v);
    const unsigned short lo = f32_to_bf16_rne(v - bf16_to_f32(hi));
    Ehi[k * CDIM + t] = hi;
    Elo[k * CDIM + t] = lo;
    float sq = v * v;
    #pragma unroll
    for (int off = 16; off >= 1; off >>= 1) sq += __shfl_xor(sq, off, 32);
    if ((t & 31) == 0) s[t >> 5] = sq;
    __syncthreads();
    if (t == 0) {
        float tot = 0.f;
        #pragma unroll
        for (int i = 0; i < 8; ++i) tot += s[i];
        esq[k] = tot;
    }
}

// ---------------------------------------------------------------------------
// Prep: hidden (NCHW) -> x rows (N x C) bf16 hi/lo + exact ||x||^2
// grid = NROWS blocks, 256 threads (one channel per thread)
// ---------------------------------------------------------------------------
__global__ void __launch_bounds__(256)
vq_prep_x(const float* __restrict__ hidden,
          unsigned short* __restrict__ Xhi,
          unsigned short* __restrict__ Xlo,
          float* __restrict__ xsq) {
    __shared__ float s[8];
    const int n  = blockIdx.x;
    const int t  = threadIdx.x;
    const int b  = n >> 10;
    const int hw = n & (HWSZ - 1);
    const float v = hidden[((b * CDIM + t) << 10) + hw];
    const unsigned short hi = f32_to_bf16_rne(v);
    const unsigned short lo = f32_to_bf16_rne(v - bf16_to_f32(hi));
    Xhi[n * CDIM + t] = hi;
    Xlo[n * CDIM + t] = lo;
    float sq = v * v;
    #pragma unroll
    for (int off = 16; off >= 1; off >>= 1) sq += __shfl_xor(sq, off, 32);
    if ((t & 31) == 0) s[t >> 5] = sq;
    __syncthreads();
    if (t == 0) {
        float tot = 0.f;
        #pragma unroll
        for (int i = 0; i < 8; ++i) tot += s[i];
        xsq[n] = tot;
    }
}

// ---------------------------------------------------------------------------
// Fused distance GEMM + argmin + gather + loss partials.
// 128 blocks x 256 threads (8 waves). Wave w owns 16 consecutive x rows.
// A panel (16x256 hi+lo) lives in registers across the whole K loop.
// B tiles (16 codes x 256 C, hi+lo) are staged in LDS per workgroup.
// Per 16-code tile: 8 K-chunks x 3 WMMAs (hi*hi + hi*lo + lo*hi) -> f32 dot.
// ---------------------------------------------------------------------------
__global__ void __launch_bounds__(256)
vq_argmin_gemm(const unsigned short* __restrict__ Xhi,
               const unsigned short* __restrict__ Xlo,
               const unsigned short* __restrict__ Ehi,
               const unsigned short* __restrict__ Elo,
               const float* __restrict__ xsq,
               const float* __restrict__ esq,
               const float* __restrict__ emb,
               const float* __restrict__ hidden,
               float* __restrict__ outQuant,
               float* __restrict__ outIdx,
               float* __restrict__ partial) {
    __shared__ unsigned short sBhi[16 * CDIM];   // 8 KB
    __shared__ unsigned short sBlo[16 * CDIM];   // 8 KB
    __shared__ int   sIdx[8][16];
    const int tid  = threadIdx.x;
    const int lane = tid & 31;
    const int wave = tid >> 5;
    const int half = lane >> 4;          // which 16-lane half
    const int l16  = lane & 15;
    const int waveGlobal = blockIdx.x * 8 + wave;
    const int rowBase    = waveGlobal * 16;

    // ---- load A panel into registers (documented 16-bit A striping) ----
    // lane<16 : K = c0+0..7 then c0+16..23 ; lane>=16 : K = c0+8..15 then c0+24..31
    BfVec ahi[8], alo[8];
    const unsigned short* arow_hi = Xhi + (rowBase + l16) * CDIM;
    const unsigned short* arow_lo = Xlo + (rowBase + l16) * CDIM;
    #pragma unroll
    for (int ck = 0; ck < 8; ++ck) {
        const int c0 = ck * 32 + half * 8;
        ahi[ck].q[0] = *(const uint4*)(arow_hi + c0);
        ahi[ck].q[1] = *(const uint4*)(arow_hi + c0 + 16);
        alo[ck].q[0] = *(const uint4*)(arow_lo + c0);
        alo[ck].q[1] = *(const uint4*)(arow_lo + c0 + 16);
    }
    // ||x||^2 for the 8 output rows this lane's accumulator slots represent
    float xs[8];
    #pragma unroll
    for (int r = 0; r < 8; ++r) xs[r] = xsq[rowBase + r + 8 * half];

    float bestD[8]; int bestI[8];
    #pragma unroll
    for (int r = 0; r < 8; ++r) { bestD[r] = 3.4e38f; bestI[r] = 0; }

    // staging map: thread t copies 32B of hi + 32B of lo
    const int sRow = tid >> 4;
    const int sCol = (tid & 15) * 16;

    for (int kt = 0; kt < KCODES / 16; ++kt) {
        const int codeBase = kt * 16;
        __syncthreads();   // protect previous tile reads
        {
            const unsigned short* gh = Ehi + (codeBase + sRow) * CDIM + sCol;
            const unsigned short* gl = Elo + (codeBase + sRow) * CDIM + sCol;
            *(uint4*)(sBhi + sRow * CDIM + sCol)     = *(const uint4*)gh;
            *(uint4*)(sBhi + sRow * CDIM + sCol + 8) = *(const uint4*)(gh + 8);
            *(uint4*)(sBlo + sRow * CDIM + sCol)     = *(const uint4*)gl;
            *(uint4*)(sBlo + sRow * CDIM + sCol + 8) = *(const uint4*)(gl + 8);
            if (kt + 1 < KCODES / 16) {   // global_prefetch_b8 hint for next tile
                __builtin_prefetch(Ehi + (codeBase + 16 + sRow) * CDIM + sCol, 0, 0);
                __builtin_prefetch(Elo + (codeBase + 16 + sRow) * CDIM + sCol, 0, 0);
            }
        }
        __syncthreads();

        v8f acc = {0.f, 0.f, 0.f, 0.f, 0.f, 0.f, 0.f, 0.f};
        #pragma unroll
        for (int ck = 0; ck < 8; ++ck) {
            // B striping: lane -> column n=l16, K = half*8 + 0..7 then half*8 + 16..23
            const int c0 = ck * 32 + half * 8;
            BfVec bhi, blo;
            const unsigned short* bh = sBhi + l16 * CDIM + c0;
            const unsigned short* bl = sBlo + l16 * CDIM + c0;
            bhi.q[0] = *(const uint4*)bh;       bhi.q[1] = *(const uint4*)(bh + 16);
            blo.q[0] = *(const uint4*)bl;       blo.q[1] = *(const uint4*)(bl + 16);
            acc = __builtin_amdgcn_wmma_f32_16x16x32_bf16(false, ahi[ck].v, false, bhi.v,
                                                          (short)0, acc, false, false);
            acc = __builtin_amdgcn_wmma_f32_16x16x32_bf16(false, ahi[ck].v, false, blo.v,
                                                          (short)0, acc, false, false);
            acc = __builtin_amdgcn_wmma_f32_16x16x32_bf16(false, alo[ck].v, false, bhi.v,
                                                          (short)0, acc, false, false);
        }
        const float eq   = esq[codeBase + l16];
        const int   code = codeBase + l16;
        #pragma unroll
        for (int r = 0; r < 8; ++r) {
            const float d = fmaf(-2.0f, acc[r], xs[r] + eq);
            if (d < bestD[r]) { bestD[r] = d; bestI[r] = code; }
        }
    }

    // ---- cross-lane argmin within each 16-lane half (same output row set) ----
    #pragma unroll
    for (int r = 0; r < 8; ++r) {
        for (int off = 1; off < 16; off <<= 1) {
            const float od = __shfl_xor(bestD[r], off, 32);
            const int   oi = __shfl_xor(bestI[r], off, 32);
            if (od < bestD[r] || (od == bestD[r] && oi < bestI[r])) {
                bestD[r] = od; bestI[r] = oi;
            }
        }
    }
    if (l16 == 0) {
        #pragma unroll
        for (int r = 0; r < 8; ++r) sIdx[wave][r + 8 * half] = bestI[r];
    }
    __syncthreads();

    // ---- epilogue: gather winning codes, write NCHW quant, loss partial ----
    const int b      = rowBase >> 10;
    const int hw     = (rowBase & (HWSZ - 1)) + l16;
    const int code   = sIdx[wave][l16];
    float lossAcc = 0.f;
    #pragma unroll 4
    for (int i = 0; i < CDIM / 2; ++i) {
        const int   c  = 2 * i + half;                     // half0 even c, half1 odd c
        const float ev = emb[code * CDIM + c];
        const int   go = ((b * CDIM + c) << 10) + hw;
        const float xv = hidden[go];
        const float df = ev - xv;
        lossAcc = fmaf(df, df, lossAcc);
        outQuant[go] = ev;                                 // forward value == quant
    }
    #pragma unroll
    for (int off = 16; off >= 1; off >>= 1) lossAcc += __shfl_xor(lossAcc, off, 32);
    if (lane == 0) partial[waveGlobal] = lossAcc;
    if (half == 0) outIdx[rowBase + l16] = (float)code;
}

// ---------------------------------------------------------------------------
// Deterministic loss finalize: sum 1024 per-wave partials in fixed order.
// ---------------------------------------------------------------------------
__global__ void __launch_bounds__(256)
vq_loss_finalize(const float* __restrict__ partial, float* __restrict__ outLoss) {
    __shared__ float s[8];
    const int t = threadIdx.x;
    float v = partial[t] + partial[t + 256] + partial[t + 512] + partial[t + 768];
    #pragma unroll
    for (int off = 16; off >= 1; off >>= 1) v += __shfl_xor(v, off, 32);
    if ((t & 31) == 0) s[t >> 5] = v;
    __syncthreads();
    if (t == 0) {
        float tot = 0.f;
        #pragma unroll
        for (int i = 0; i < 8; ++i) tot += s[i];
        // loss = mean(d^2) + BETA*mean(d^2), BETA = 0.25 (stop_grad is forward no-op)
        *outLoss = 1.25f * tot / ((float)NROWS * (float)CDIM);
    }
}

// ---------------------------------------------------------------------------
// Launch. d_in[0]=hidden_state f32 [16,256,32,32]; d_in[1]=emb_weight f32 [8192,256]
// d_out = [quant 16*256*32*32 | loss 1 | indices 16*1024] as float.
// ---------------------------------------------------------------------------
extern "C" void kernel_launch(void* const* d_in, const int* in_sizes, int n_in,
                              void* d_out, int out_size, void* d_ws, size_t ws_size,
                              hipStream_t stream) {
    const float* hidden = (const float*)d_in[0];
    const float* emb    = (const float*)d_in[1];

    float* out      = (float*)d_out;
    float* outQuant = out;                        // 4,194,304 floats
    float* outLoss  = out + NROWS * CDIM;         // 1 float
    float* outIdx   = out + NROWS * CDIM + 1;     // 16,384 floats

    // workspace carving (~25.3 MB)
    unsigned short* Ehi = (unsigned short*)d_ws;
    unsigned short* Elo = Ehi + (size_t)KCODES * CDIM;
    unsigned short* Xhi = Elo + (size_t)KCODES * CDIM;
    unsigned short* Xlo = Xhi + (size_t)NROWS * CDIM;
    float* esq     = (float*)(Xlo + (size_t)NROWS * CDIM);
    float* xsq     = esq + KCODES;
    float* partial = xsq + NROWS;                 // 1024 floats

    vq_prep_codes<<<KCODES, 256, 0, stream>>>(emb, Ehi, Elo, esq);
    vq_prep_x   <<<NROWS,  256, 0, stream>>>(hidden, Xhi, Xlo, xsq);
    vq_argmin_gemm<<<NROWS / 128, 256, 0, stream>>>(Xhi, Xlo, Ehi, Elo, xsq, esq,
                                                    emb, hidden, outQuant, outIdx, partial);
    vq_loss_finalize<<<1, 256, 0, stream>>>(partial, outLoss);
}
